// TCRgnnEdge_22720376996120
// MI455X (gfx1250) — compile-verified, matrix operand order checked by reading, not run on hardware
//
#include <hip/hip_runtime.h>

typedef __attribute__((ext_vector_type(16))) __bf16 v16bf;
typedef __attribute__((ext_vector_type(8)))  float  v8f;

#define N_NODES    100000
#define N_EDGES    1600000
#define DIM        64
#define NUM_GRAPHS 1000

__device__ __forceinline__ __bf16 f2bf(float f) { return (__bf16)f; }

__device__ __forceinline__ void atomAddF(float* p, float v) {
    // hardware global_atomic_add_f32 (resolves at L2) — no CAS loop
    unsafeAtomicAdd(p, v);
}

// ---------------------------------------------------------------------------
// Zero-fill scratch (harness poisons ws; we must establish zeros ourselves).
// ---------------------------------------------------------------------------
__global__ void fill_zero_kernel(float* __restrict__ p, int n) {
    int i = blockIdx.x * blockDim.x + threadIdx.x;
    if (i < n) p[i] = 0.0f;
}

// ---------------------------------------------------------------------------
// Pre-swizzle the four 64x64 MLP weight matrices into per-lane WMMA
// B-fragment layout (bf16). Fragment f = kb*4+nb holds, for lane = hi*16+nl,
// elements j: j<8  -> W[kb*32 +      hi*8 + j      ][nb*16+nl]
//             j>=8 -> W[kb*32 + 16 + hi*8 + (j-8) ][nb*16+nl]
// so a GEMM lane loads its whole v16bf fragment as one aligned 32B vector.
// ---------------------------------------------------------------------------
__global__ void swizzle_weights_kernel(const float* __restrict__ a,
                                       const float* __restrict__ b,
                                       const float* __restrict__ c,
                                       const float* __restrict__ d,
                                       __bf16* __restrict__ out) {
    int i = blockIdx.x * blockDim.x + threadIdx.x;   // 0 .. 4*4096-1
    if (i >= 4 * DIM * DIM) return;
    int m = i >> 12;                 // which matrix
    int o = i & 4095;                // element within matrix
    int f    = o >> 9;               // fragment 0..7  (kb*4+nb)
    int rem  = o & 511;
    int lane = rem >> 4;             // 0..31
    int j    = rem & 15;             // 0..15
    int kb = f >> 2, nb = f & 3;
    int hi = lane >> 4, nl = lane & 15;
    int half = j >> 3, jj = j & 7;
    int k = kb * 32 + half * 16 + hi * 8 + jj;
    int n = nb * 16 + nl;
    const float* s = (m == 0) ? a : (m == 1) ? b : (m == 2) ? c : d;
    out[i] = f2bf(s[k * DIM + n]);
}

// ---------------------------------------------------------------------------
// Edge message + scatter:  agg[dst] += relu(x[src] + ea*We + be)
// 16 lanes per edge -> one fully-coalesced 256B line per gather,
// FP32 hardware atomics for the scatter.
// ---------------------------------------------------------------------------
__global__ void edge_scatter_kernel(const float* __restrict__ x,
                                    const int*   __restrict__ src,
                                    const int*   __restrict__ dst,
                                    const float* __restrict__ eattr,
                                    const float* __restrict__ We,
                                    const float* __restrict__ be,
                                    float* __restrict__ agg) {
    int t = blockIdx.x * blockDim.x + threadIdx.x;
    if (t >= N_EDGES * 16) return;
    int e = t >> 4;
    int q = (t & 15) << 2;                 // feature offset (0..60)
    int s = src[e], d = dst[e];
    float ea = eattr[e];
    const float4 xv = *(const float4*)(x  + (size_t)s * DIM + q);
    const float4 wv = *(const float4*)(We + q);
    const float4 bv = *(const float4*)(be + q);
    float4 m;
    m.x = fmaxf(fmaf(ea, wv.x, xv.x) + bv.x, 0.0f);
    m.y = fmaxf(fmaf(ea, wv.y, xv.y) + bv.y, 0.0f);
    m.z = fmaxf(fmaf(ea, wv.z, xv.z) + bv.z, 0.0f);
    m.w = fmaxf(fmaf(ea, wv.w, xv.w) + bv.w, 0.0f);
    float* ap = agg + (size_t)d * DIM + q;
    atomAddF(ap + 0, m.x);
    atomAddF(ap + 1, m.y);
    atomAddF(ap + 2, m.z);
    atomAddF(ap + 3, m.w);
}

// ---------------------------------------------------------------------------
// GIN MLP:  out = act( relu( (x+agg) @ Wa + ba ) @ Wb + bb )
// 128 threads = 4 wave32; each wave computes a 16x64 tile with
// 8 x v_wmma_f32_16x16x32_bf16 per GEMM stage. Weight fragments come
// pre-swizzled from global (L1/L2-resident) as single v16bf loads.
// ---------------------------------------------------------------------------
__global__ __launch_bounds__(128)
void gin_mlp_kernel(const float*  __restrict__ xin,
                    const float*  __restrict__ agg,
                    const v16bf*  __restrict__ WaF,   // 8 frags x 32 lanes
                    const float*  __restrict__ ba,
                    const v16bf*  __restrict__ WbF,
                    const float*  __restrict__ bb,
                    float* __restrict__ out,
                    int nrows, int relu_out) {
    __shared__ float sH[4][16][68];           // pad 68: conflict-free + 16B aligned rows

    const int tid  = threadIdx.x;
    const int wave = tid >> 5;
    const int lane = tid & 31;
    const int hi   = lane >> 4;               // K-half selector (ISA A/B layout)
    const int nl   = lane & 15;               // row (A) / column (B,C,D) index
    const int rowBase = blockIdx.x * 64 + wave * 16;
    const int myRow   = rowBase + nl;

    // ---- A fragments for GEMM1 from (x + agg), per ISA 16-bit A 16x32 layout
    v16bf afrag[2];
    #pragma unroll
    for (int kb = 0; kb < 2; ++kb) {
        const int k0 = kb * 32;
        if (myRow < nrows) {
            const float* rp = xin + (size_t)myRow * DIM;
            const float* gp = agg + (size_t)myRow * DIM;
            const float4* xa = (const float4*)(rp + k0 + hi * 8);       // 8 floats
            const float4* ga = (const float4*)(gp + k0 + hi * 8);
            const float4* xb = (const float4*)(rp + k0 + 16 + hi * 8);  // 8 floats
            const float4* gb = (const float4*)(gp + k0 + 16 + hi * 8);
            float4 v0 = xa[0], v1 = xa[1], g0 = ga[0], g1 = ga[1];
            float4 u0 = xb[0], u1 = xb[1], h0 = gb[0], h1 = gb[1];
            afrag[kb][0]  = f2bf(v0.x + g0.x); afrag[kb][1]  = f2bf(v0.y + g0.y);
            afrag[kb][2]  = f2bf(v0.z + g0.z); afrag[kb][3]  = f2bf(v0.w + g0.w);
            afrag[kb][4]  = f2bf(v1.x + g1.x); afrag[kb][5]  = f2bf(v1.y + g1.y);
            afrag[kb][6]  = f2bf(v1.z + g1.z); afrag[kb][7]  = f2bf(v1.w + g1.w);
            afrag[kb][8]  = f2bf(u0.x + h0.x); afrag[kb][9]  = f2bf(u0.y + h0.y);
            afrag[kb][10] = f2bf(u0.z + h0.z); afrag[kb][11] = f2bf(u0.w + h0.w);
            afrag[kb][12] = f2bf(u1.x + h1.x); afrag[kb][13] = f2bf(u1.y + h1.y);
            afrag[kb][14] = f2bf(u1.z + h1.z); afrag[kb][15] = f2bf(u1.w + h1.w);
        } else {
            #pragma unroll
            for (int j = 0; j < 16; ++j) afrag[kb][j] = f2bf(0.0f);
        }
    }

    // ---- GEMM1 + bias + relu -> sH  (one acc + one bfrag live at a time)
    #pragma unroll 1
    for (int nb = 0; nb < 4; ++nb) {
        const int n = nb * 16 + nl;
        v8f acc = {};
        #pragma unroll
        for (int kb = 0; kb < 2; ++kb) {
            const v16bf bfrag = WaF[(kb * 4 + nb) * 32 + lane];   // 2x b128
            acc = __builtin_amdgcn_wmma_f32_16x16x32_bf16(
                      false, afrag[kb], false, bfrag, (short)0, acc, false, false);
        }
        const float bias = ba[n];
        #pragma unroll
        for (int v = 0; v < 8; ++v)
            sH[wave][hi * 8 + v][n] = fmaxf(acc[v] + bias, 0.0f);
    }
    __syncthreads();

    // ---- rebuild A fragments from sH (C column-striped -> A row-striped)
    #pragma unroll
    for (int kb = 0; kb < 2; ++kb) {
        const int k0 = kb * 32;
        const float4* pa = (const float4*)(&sH[wave][nl][k0 + hi * 8]);
        const float4* pb = (const float4*)(&sH[wave][nl][k0 + 16 + hi * 8]);
        float4 v0 = pa[0], v1 = pa[1], u0 = pb[0], u1 = pb[1];
        afrag[kb][0]  = f2bf(v0.x); afrag[kb][1]  = f2bf(v0.y);
        afrag[kb][2]  = f2bf(v0.z); afrag[kb][3]  = f2bf(v0.w);
        afrag[kb][4]  = f2bf(v1.x); afrag[kb][5]  = f2bf(v1.y);
        afrag[kb][6]  = f2bf(v1.z); afrag[kb][7]  = f2bf(v1.w);
        afrag[kb][8]  = f2bf(u0.x); afrag[kb][9]  = f2bf(u0.y);
        afrag[kb][10] = f2bf(u0.z); afrag[kb][11] = f2bf(u0.w);
        afrag[kb][12] = f2bf(u1.x); afrag[kb][13] = f2bf(u1.y);
        afrag[kb][14] = f2bf(u1.z); afrag[kb][15] = f2bf(u1.w);
    }

    // ---- GEMM2 + bias (+ optional relu) -> out
    #pragma unroll 1
    for (int nb = 0; nb < 4; ++nb) {
        const int n = nb * 16 + nl;
        v8f acc = {};
        #pragma unroll
        for (int kb = 0; kb < 2; ++kb) {
            const v16bf bfrag = WbF[(kb * 4 + nb) * 32 + lane];
            acc = __builtin_amdgcn_wmma_f32_16x16x32_bf16(
                      false, afrag[kb], false, bfrag, (short)0, acc, false, false);
        }
        const float bias = bb[n];
        #pragma unroll
        for (int v = 0; v < 8; ++v) {
            int r = rowBase + hi * 8 + v;
            if (r < nrows) {
                float h = acc[v] + bias;
                out[(size_t)r * DIM + n] = relu_out ? fmaxf(h, 0.0f) : h;
            }
        }
    }
}

// ---------------------------------------------------------------------------
// Segment-sum pooling: sums[batch[n]] += h[n]; cnts[batch[n]] += 1
// ---------------------------------------------------------------------------
__global__ void pool_scatter_kernel(const float* __restrict__ h,
                                    const int*   __restrict__ batch,
                                    float* __restrict__ sums,
                                    float* __restrict__ cnts) {
    int t = blockIdx.x * blockDim.x + threadIdx.x;
    if (t >= N_NODES * 16) return;
    int nid = t >> 4;
    int q   = (t & 15) << 2;
    int g   = batch[nid];
    const float4 hv = *(const float4*)(h + (size_t)nid * DIM + q);
    float* sp = sums + (size_t)g * DIM + q;
    atomAddF(sp + 0, hv.x);
    atomAddF(sp + 1, hv.y);
    atomAddF(sp + 2, hv.z);
    atomAddF(sp + 3, hv.w);
    if (q == 0) atomAddF(&cnts[g], 1.0f);
}

// ---------------------------------------------------------------------------
// Per-graph classifier: relu(pooled @ Wc1 + bc1) @ Wc2 + bc2  (tiny)
// ---------------------------------------------------------------------------
__global__ __launch_bounds__(64)
void classify_kernel(const float* __restrict__ sums,
                     const float* __restrict__ cnts,
                     const float* __restrict__ Wc1,
                     const float* __restrict__ bc1,
                     const float* __restrict__ Wc2,
                     const float* __restrict__ bc2,
                     float* __restrict__ out) {
    __shared__ float sp[DIM];
    __shared__ float sy[DIM];
    const int g = blockIdx.x, j = threadIdx.x;
    const float c = fmaxf(cnts[g], 1.0f);
    sp[j] = sums[(size_t)g * DIM + j] / c;
    __syncthreads();
    float acc = bc1[j];
    #pragma unroll 1
    for (int d = 0; d < DIM; ++d) acc = fmaf(sp[d], Wc1[d * DIM + j], acc);
    sy[j] = fmaxf(acc, 0.0f) * Wc2[j];
    __syncthreads();
    for (int s = 32; s > 0; s >>= 1) {
        if (j < s) sy[j] += sy[j + s];
        __syncthreads();
    }
    if (j == 0) out[g] = sy[0] + bc2[0];
}

// ---------------------------------------------------------------------------
extern "C" void kernel_launch(void* const* d_in, const int* in_sizes, int n_in,
                              void* d_out, int out_size, void* d_ws, size_t ws_size,
                              hipStream_t stream) {
    const float* x     = (const float*)d_in[0];
    const int*   ei    = (const int*)  d_in[1];   // [2, E]
    const float* eattr = (const float*)d_in[2];
    const int*   batch = (const int*)  d_in[3];
    const float* W1a = (const float*)d_in[4],  *b1a = (const float*)d_in[5];
    const float* W1b = (const float*)d_in[6],  *b1b = (const float*)d_in[7];
    const float* We1 = (const float*)d_in[8],  *be1 = (const float*)d_in[9];
    const float* W2a = (const float*)d_in[10], *b2a = (const float*)d_in[11];
    const float* W2b = (const float*)d_in[12], *b2b = (const float*)d_in[13];
    const float* We2 = (const float*)d_in[14], *be2 = (const float*)d_in[15];
    const float* Wc1 = (const float*)d_in[16], *bc1 = (const float*)d_in[17];
    const float* Wc2 = (const float*)d_in[18], *bc2 = (const float*)d_in[19];
    float* out = (float*)d_out;

    const int* srcI = ei;
    const int* dstI = ei + N_EDGES;

    // Workspace layout — bf16 fragment buffer first so it inherits the
    // allocation's 256B alignment (needed for v16bf 32B vector loads).
    const size_t nfeat = (size_t)N_NODES * DIM;
    char* ws = (char*)d_ws;
    __bf16* wfrag = (__bf16*)ws;  ws += (size_t)4 * DIM * DIM * sizeof(__bf16);
    float*  agg   = (float*)ws;   ws += nfeat * sizeof(float);
    float*  h1    = (float*)ws;   ws += nfeat * sizeof(float);
    float*  h2    = (float*)ws;   ws += nfeat * sizeof(float);
    float*  sums  = (float*)ws;   ws += (size_t)NUM_GRAPHS * DIM * sizeof(float);
    float*  cnts  = (float*)ws;   ws += (size_t)NUM_GRAPHS * sizeof(float);

    const v16bf* W1aF = (const v16bf*)(wfrag);
    const v16bf* W1bF = (const v16bf*)(wfrag + 4096);
    const v16bf* W2aF = (const v16bf*)(wfrag + 8192);
    const v16bf* W2bF = (const v16bf*)(wfrag + 12288);

    const int featInt   = (int)nfeat;
    const int edgeThr   = N_EDGES * 16;
    const int nodeThr   = N_NODES * 16;
    const int mlpBlocks = (N_NODES + 63) / 64;

    // weights -> pre-swizzled bf16 fragments
    swizzle_weights_kernel<<<(4 * DIM * DIM + 255) / 256, 256, 0, stream>>>(
        W1a, W1b, W2a, W2b, wfrag);

    // ---- conv1
    fill_zero_kernel<<<(featInt + 255) / 256, 256, 0, stream>>>(agg, featInt);
    edge_scatter_kernel<<<(edgeThr + 255) / 256, 256, 0, stream>>>(
        x, srcI, dstI, eattr, We1, be1, agg);
    gin_mlp_kernel<<<mlpBlocks, 128, 0, stream>>>(
        x, agg, W1aF, b1a, W1bF, b1b, h1, N_NODES, /*relu_out=*/1);

    // ---- conv2
    fill_zero_kernel<<<(featInt + 255) / 256, 256, 0, stream>>>(agg, featInt);
    edge_scatter_kernel<<<(edgeThr + 255) / 256, 256, 0, stream>>>(
        h1, srcI, dstI, eattr, We2, be2, agg);
    gin_mlp_kernel<<<mlpBlocks, 128, 0, stream>>>(
        h1, agg, W2aF, b2a, W2bF, b2b, h2, N_NODES, /*relu_out=*/1);

    // ---- pooling + classifier (sums and cnts contiguous -> one fill)
    fill_zero_kernel<<<(NUM_GRAPHS * DIM + NUM_GRAPHS + 255) / 256, 256, 0, stream>>>(
        sums, NUM_GRAPHS * DIM + NUM_GRAPHS);
    pool_scatter_kernel<<<(nodeThr + 255) / 256, 256, 0, stream>>>(
        h2, batch, sums, cnts);
    classify_kernel<<<NUM_GRAPHS, 64, 0, stream>>>(
        sums, cnts, Wc1, bc1, Wc2, bc2, out);
}
